// BlockWithCrossAtten_80882824118886
// MI455X (gfx1250) — matrix operand order repaired and checked by reading, not compiled, not run on hardware
//
#include <hip/hip_runtime.h>
#include <math.h>

typedef __attribute__((ext_vector_type(16))) __bf16 v16bf;
typedef __attribute__((ext_vector_type(8)))  float  v8f;

#define CC   1024
#define HH   16
#define DHD  64
#define TT   2048
#define BB   4
#define SS   64
#define BT   (BB*TT)    /* 8192 rows of x */
#define BSR  (BB*SS)    /* 256 rows of z  */
#define FC   4096

// ---------------- helpers ----------------

static __device__ __forceinline__ unsigned short f2bf(float f) {
  unsigned int u = __float_as_uint(f);
  u += 0x7FFFu + ((u >> 16) & 1u);          // round-to-nearest-even
  return (unsigned short)(u >> 16);
}

static __device__ __forceinline__ v16bf frag2(const unsigned short* p0, const unsigned short* p1) {
  union { uint4 u[2]; v16bf f; } x;
  x.u[0] = *(const uint4*)p0;
  x.u[1] = *(const uint4*)p1;
  return x.f;
}
static __device__ __forceinline__ v16bf frag32(const unsigned short* p) { return frag2(p, p + 8); }

static __device__ __forceinline__ v8f vzero() {
  v8f v = {0.f,0.f,0.f,0.f,0.f,0.f,0.f,0.f};
  return v;
}
static __device__ __forceinline__ v8f wmma_bf16(v16bf a, v16bf b, v8f c) {
  // D = A(16x32 bf16) x B(32x16 bf16) + C(16x16 f32)
  return __builtin_amdgcn_wmma_f32_16x16x32_bf16(false, a, false, b, (short)0, c, false, false);
}
static __device__ __forceinline__ float gelu_f(float x) {
  float x3 = x * x * x;
  return 0.5f * x * (1.f + tanhf(0.7978845608028654f * (x + 0.044715f * x3)));
}

// ---- CDNA5 async global->LDS DMA (ASYNCcnt-tracked), per ISA ch.10 / 08_async_tensor ----
// LDS byte address = low 32 bits of the generic pointer (flat aperture truncation).
static __device__ __forceinline__ void async_b128(const unsigned short* g, unsigned short* l) {
  asm volatile("global_load_async_to_lds_b128 %0, %1, off"
               :: "v"((unsigned)(unsigned long long)l),
                  "v"((unsigned long long)g)
               : "memory");
}
static __device__ __forceinline__ void wait_async_le6() {
  asm volatile("s_wait_asynccnt 6" ::: "memory");
}
static __device__ __forceinline__ void wait_async_0() {
  asm volatile("s_wait_asynccnt 0" ::: "memory");
}

// ---------------- fp32 -> bf16 conversion ----------------

__global__ __launch_bounds__(256) void k_convert(const float* __restrict__ s,
                                                 unsigned short* __restrict__ d, int n) {
  for (int i = blockIdx.x * blockDim.x + threadIdx.x; i < n; i += gridDim.x * blockDim.x)
    d[i] = f2bf(s[i]);
}

// ---------------- LayerNorm (fp32 in, bf16 out), one block per row of C=1024 ----------------

__global__ __launch_bounds__(256) void k_layernorm(const float* __restrict__ x,
                                                   const float* __restrict__ g,
                                                   const float* __restrict__ b,
                                                   unsigned short* __restrict__ o) {
  __shared__ float red[2][8];
  const int row = blockIdx.x, tid = threadIdx.x;
  const float4 v = ((const float4*)(x + (size_t)row * CC))[tid];
  float s = v.x + v.y + v.z + v.w;
  float q = v.x*v.x + v.y*v.y + v.z*v.z + v.w*v.w;
  #pragma unroll
  for (int m = 16; m; m >>= 1) { s += __shfl_xor(s, m, 32); q += __shfl_xor(q, m, 32); }
  if ((tid & 31) == 0) { red[0][tid >> 5] = s; red[1][tid >> 5] = q; }
  __syncthreads();
  s = 0.f; q = 0.f;
  #pragma unroll
  for (int i = 0; i < 8; ++i) { s += red[0][i]; q += red[1][i]; }
  const float mu   = s * (1.f / CC);
  const float rinv = rsqrtf(q * (1.f / CC) - mu * mu + 1e-5f);
  const float4 gw = ((const float4*)g)[tid];
  const float4 gb = ((const float4*)b)[tid];
  unsigned short* op = o + (size_t)row * CC + tid * 4;
  op[0] = f2bf((v.x - mu) * rinv * gw.x + gb.x);
  op[1] = f2bf((v.y - mu) * rinv * gw.y + gb.y);
  op[2] = f2bf((v.z - mu) * rinv * gw.z + gb.z);
  op[3] = f2bf((v.w - mu) * rinv * gw.w + gb.w);
}

// ---------------- WMMA GEMM: out = A[M,K] @ W[N,K]^T + bias ----------------
// mode 0: bf16 out = acc+bias ; mode 1: bf16 out = gelu(acc+bias)
// mode 2: f32 out  = residual + acc + bias
// Block tile 128x64, K-step 64; double-buffered LDS fed by async global->LDS DMA.

__global__ __launch_bounds__(256) void k_gemm(const unsigned short* __restrict__ A,
                                              const unsigned short* __restrict__ W,
                                              const float* __restrict__ bias,
                                              const float* __restrict__ residual,
                                              void* __restrict__ out,
                                              int M, int N, int K, int mode) {
  __shared__ __align__(16) unsigned short sA[2][128][64];   // 32 KB
  __shared__ __align__(16) unsigned short sB[2][64][64];    // 16 KB
  const int tid  = threadIdx.x;
  const int lane = tid & 31, wave = tid >> 5;
  const int half = lane >> 4, l16 = lane & 15;
  const int wm = (wave >> 1) * 32, wn = (wave & 1) * 32;
  const int m0 = blockIdx.y * 128, n0 = blockIdx.x * 64;

  v8f acc[2][2] = { { vzero(), vzero() }, { vzero(), vzero() } };

  auto issue = [&](int buf, int k0) {
    #pragma unroll
    for (int i = 0; i < 4; ++i) {                 // A tile: 1024 x 16B chunks
      int c = tid + 256 * i;
      int r = c >> 3, cc = (c & 7) * 8;
      async_b128(&A[(size_t)(m0 + r) * K + k0 + cc], &sA[buf][r][cc]);
    }
    #pragma unroll
    for (int i = 0; i < 2; ++i) {                 // W tile: 512 x 16B chunks
      int c = tid + 256 * i;
      int r = c >> 3, cc = (c & 7) * 8;
      async_b128(&W[(size_t)(n0 + r) * K + k0 + cc], &sB[buf][r][cc]);
    }
  };

  issue(0, 0);
  const int nk = K >> 6;
  for (int kc = 0; kc < nk; ++kc) {
    const int buf  = kc & 1;
    const bool more = (kc + 1 < nk);
    if (more) issue(buf ^ 1, (kc + 1) << 6);      // prefetch next slice into other buffer
    if (more) wait_async_le6();                   // 6 newer ops outstanding -> this buf done
    else      wait_async_0();
    __syncthreads();

    #pragma unroll
    for (int kk = 0; kk < 64; kk += 32) {
      v16bf af[2], bfr[2];
      #pragma unroll
      for (int mt = 0; mt < 2; ++mt) {
        const unsigned short* p = &sA[buf][wm + mt * 16 + l16][kk + half * 16];
        af[mt] = frag2(p, p + 8);                 // K: half*16..+15
      }
      #pragma unroll
      for (int nt = 0; nt < 2; ++nt)
        bfr[nt] = frag32(&sB[buf][wn + nt * 16 + l16][kk + half * 16]);
      #pragma unroll
      for (int mt = 0; mt < 2; ++mt)
        #pragma unroll
        for (int nt = 0; nt < 2; ++nt)
          acc[mt][nt] = wmma_bf16(af[mt], bfr[nt], acc[mt][nt]);
    }
    __syncthreads();
  }

  #pragma unroll
  for (int mt = 0; mt < 2; ++mt)
    #pragma unroll
    for (int nt = 0; nt < 2; ++nt)
      #pragma unroll
      for (int r = 0; r < 8; ++r) {
        const int row = m0 + wm + mt * 16 + half * 8 + r;
        const int col = n0 + wn + nt * 16 + l16;
        float v = acc[mt][nt][r] + bias[col];
        if (mode == 1) v = gelu_f(v);
        const size_t idx = (size_t)row * N + col;
        if (mode == 2) ((float*)out)[idx] = residual[idx] + v;
        else           ((unsigned short*)out)[idx] = f2bf(v);
      }
}

// ---------------- Flash attention (causal self-attn and cross-attn) ----------------
// Block: 128 threads (4 waves), 64 query rows per block, stream 64-key tiles.
// q/k/v are bf16 with row stride *_rs and column offset *_co; head dim 64; out row stride CC.

__global__ __launch_bounds__(128) void k_flash_attn(
    const unsigned short* __restrict__ q, int q_rs, int q_co,
    const unsigned short* __restrict__ k, int k_rs, int k_co,
    const unsigned short* __restrict__ v, int v_rs, int v_co,
    unsigned short* __restrict__ o, int Tq, int Tkv, int causal) {
  __shared__ __align__(16) unsigned short sK[64][64];       // [s][d]  8 KB
  __shared__ __align__(16) unsigned short sVt[64][64];      // [d][s]  8 KB (transposed)
  __shared__ __align__(16) unsigned short sP[4][16][64];    // per-wave P tile, 8 KB

  const int b = blockIdx.z, h = blockIdx.y, q0 = blockIdx.x * 64;
  const int tid  = threadIdx.x;
  const int lane = tid & 31, w = tid >> 5;
  const int half = lane >> 4, l16 = lane & 15;

  const unsigned short* qp = q + (size_t)b * Tq  * q_rs + q_co + h * DHD;
  const unsigned short* kp = k + (size_t)b * Tkv * k_rs + k_co + h * DHD;
  const unsigned short* vp = v + (size_t)b * Tkv * v_rs + v_co + h * DHD;
  unsigned short*       op = o + (size_t)b * Tq  * CC   + h * DHD;

  // Q fragments (16 rows x 64 d), loaded once
  v16bf qf[2];
  {
    const unsigned short* pr = qp + (size_t)(q0 + w * 16 + l16) * q_rs;
    #pragma unroll
    for (int i = 0; i < 2; ++i) {
      const unsigned short* p = pr + i * 32 + half * 16;
      qf[i] = frag2(p, p + 8);
    }
  }

  v8f accO[4] = { vzero(), vzero(), vzero(), vzero() };
  float mI[8], lI[8];
  #pragma unroll
  for (int r = 0; r < 8; ++r) { mI[r] = -1e30f; lI[r] = 0.f; }

  const int nTiles = causal ? (blockIdx.x + 1) : (Tkv / 64);

  for (int st = 0; st < nTiles; ++st) {
    const int s0 = st * 64;
    // K tile: async DMA straight into LDS; V tile: via VGPRs with transpose scatter
    #pragma unroll
    for (int i = 0; i < 4; ++i) {
      int c = tid + 128 * i;
      int r = c >> 3, cc = (c & 7) * 8;
      async_b128(&kp[(size_t)(s0 + r) * k_rs + cc], &sK[r][cc]);
      union { uint4 u; unsigned short s[8]; } t;
      t.u = *(const uint4*)&vp[(size_t)(s0 + r) * v_rs + cc];
      #pragma unroll
      for (int j = 0; j < 8; ++j) sVt[cc + j][r] = t.s[j];
    }
    wait_async_0();
    __syncthreads();

    // S = Q K^T  (16 x 64 per wave)
    v8f sc[4] = { vzero(), vzero(), vzero(), vzero() };
    #pragma unroll
    for (int nt = 0; nt < 4; ++nt)
      #pragma unroll
      for (int i = 0; i < 2; ++i)
        sc[nt] = wmma_bf16(qf[i], frag32(&sK[nt * 16 + l16][i * 32 + half * 16]), sc[nt]);

    const int mrow = q0 + w * 16 + half * 8;
    #pragma unroll
    for (int nt = 0; nt < 4; ++nt)
      #pragma unroll
      for (int r = 0; r < 8; ++r) {
        float vv = sc[nt][r] * 0.125f;          // 1/sqrt(64)
        if (causal && (s0 + nt * 16 + l16) > (mrow + r)) vv = -1e30f;
        sc[nt][r] = vv;
      }

    // online softmax per row (row = half*8 + r, 16 cols per lane-group)
    #pragma unroll
    for (int r = 0; r < 8; ++r) {
      float rmax = fmaxf(fmaxf(sc[0][r], sc[1][r]), fmaxf(sc[2][r], sc[3][r]));
      #pragma unroll
      for (int m = 1; m < 16; m <<= 1) rmax = fmaxf(rmax, __shfl_xor(rmax, m, 32));
      const float mnew  = fmaxf(mI[r], rmax);
      const float alpha = __expf(mI[r] - mnew);
      float rsum = 0.f;
      #pragma unroll
      for (int nt = 0; nt < 4; ++nt) {
        float p = __expf(sc[nt][r] - mnew);
        sc[nt][r] = p;
        rsum += p;
      }
      #pragma unroll
      for (int m = 1; m < 16; m <<= 1) rsum += __shfl_xor(rsum, m, 32);
      lI[r] = lI[r] * alpha + rsum;
      mI[r] = mnew;
      #pragma unroll
      for (int dt = 0; dt < 4; ++dt) accO[dt][r] *= alpha;
    }

    // P -> bf16 via per-wave LDS region, reload as A fragments
    #pragma unroll
    for (int nt = 0; nt < 4; ++nt)
      #pragma unroll
      for (int r = 0; r < 8; ++r)
        sP[w][half * 8 + r][nt * 16 + l16] = f2bf(sc[nt][r]);

    v16bf pf[2];
    #pragma unroll
    for (int i = 0; i < 2; ++i) {
      const unsigned short* p = &sP[w][l16][i * 32 + half * 16];
      pf[i] = frag2(p, p + 8);
    }
    #pragma unroll
    for (int dt = 0; dt < 4; ++dt)
      #pragma unroll
      for (int i = 0; i < 2; ++i)
        accO[dt] = wmma_bf16(pf[i], frag32(&sVt[dt * 16 + l16][i * 32 + half * 16]), accO[dt]);

    __syncthreads();
  }

  #pragma unroll
  for (int dt = 0; dt < 4; ++dt)
    #pragma unroll
    for (int r = 0; r < 8; ++r) {
      const float ov = accO[dt][r] / lI[r];
      op[(size_t)(q0 + w * 16 + half * 8 + r) * CC + dt * 16 + l16] = f2bf(ov);
    }
}

// ---------------- host orchestration ----------------

extern "C" void kernel_launch(void* const* d_in, const int* in_sizes, int n_in,
                              void* d_out, int out_size, void* d_ws, size_t ws_size,
                              hipStream_t stream) {
  (void)in_sizes; (void)n_in; (void)out_size; (void)ws_size;

  const float* x       = (const float*)d_in[0];
  const float* z       = (const float*)d_in[1];
  const float* ln1_w   = (const float*)d_in[2];
  const float* ln1_b   = (const float*)d_in[3];
  const float* qkv_w   = (const float*)d_in[4];
  const float* qkv_b   = (const float*)d_in[5];
  const float* aproj_w = (const float*)d_in[6];
  const float* aproj_b = (const float*)d_in[7];
  const float* lnc_w   = (const float*)d_in[8];
  const float* lnc_b   = (const float*)d_in[9];
  const float* cq_w    = (const float*)d_in[10];
  const float* cq_b    = (const float*)d_in[11];
  const float* ckv_w   = (const float*)d_in[12];
  const float* ckv_b   = (const float*)d_in[13];
  const float* cproj_w = (const float*)d_in[14];
  const float* cproj_b = (const float*)d_in[15];
  const float* ln2_w   = (const float*)d_in[16];
  const float* ln2_b   = (const float*)d_in[17];
  const float* fc_w    = (const float*)d_in[18];
  const float* fc_b    = (const float*)d_in[19];
  const float* mproj_w = (const float*)d_in[20];
  const float* mproj_b = (const float*)d_in[21];
  float* out = (float*)d_out;

  char* base = (char*)d_ws;
  size_t off = 0;
  auto allocB = [&](size_t bytes) -> void* {
    void* p = base + off;
    off = (off + bytes + 255) & ~(size_t)255;
    return p;
  };

  unsigned short* wb_qkv   = (unsigned short*)allocB((size_t)3 * CC * CC * 2);
  unsigned short* wb_aproj = (unsigned short*)allocB((size_t)CC * CC * 2);
  unsigned short* wb_cq    = (unsigned short*)allocB((size_t)CC * CC * 2);
  unsigned short* wb_ckv   = (unsigned short*)allocB((size_t)2 * CC * CC * 2);
  unsigned short* wb_cproj = (unsigned short*)allocB((size_t)CC * CC * 2);
  unsigned short* wb_fc    = (unsigned short*)allocB((size_t)FC * CC * 2);
  unsigned short* wb_mproj = (unsigned short*)allocB((size_t)CC * FC * 2);
  unsigned short* hb       = (unsigned short*)allocB((size_t)BT * CC * 2);
  unsigned short* qkvb     = (unsigned short*)allocB((size_t)BT * 3 * CC * 2);
  unsigned short* yb       = (unsigned short*)allocB((size_t)BT * CC * 2);
  unsigned short* qb       = (unsigned short*)allocB((size_t)BT * CC * 2);
  unsigned short* zb       = (unsigned short*)allocB((size_t)BSR * CC * 2);
  unsigned short* kvb      = (unsigned short*)allocB((size_t)BSR * 2 * CC * 2);
  unsigned short* fcb      = (unsigned short*)allocB((size_t)BT * FC * 2);
  float*          x1       = (float*)allocB((size_t)BT * CC * 4);
  float*          x2       = (float*)allocB((size_t)BT * CC * 4);

  auto conv = [&](const float* s, unsigned short* d, size_t n) {
    k_convert<<<dim3(1024), dim3(256), 0, stream>>>(s, d, (int)n);
  };
  conv(qkv_w,   wb_qkv,   (size_t)3 * CC * CC);
  conv(aproj_w, wb_aproj, (size_t)CC * CC);
  conv(cq_w,    wb_cq,    (size_t)CC * CC);
  conv(ckv_w,   wb_ckv,   (size_t)2 * CC * CC);
  conv(cproj_w, wb_cproj, (size_t)CC * CC);
  conv(fc_w,    wb_fc,    (size_t)FC * CC);
  conv(mproj_w, wb_mproj, (size_t)CC * FC);
  conv(z,       zb,       (size_t)BSR * CC);

  // --- causal self-attention ---
  k_layernorm<<<dim3(BT), dim3(256), 0, stream>>>(x, ln1_w, ln1_b, hb);
  k_gemm<<<dim3(3 * CC / 64, BT / 128), dim3(256), 0, stream>>>(
      hb, wb_qkv, qkv_b, nullptr, qkvb, BT, 3 * CC, CC, 0);
  k_flash_attn<<<dim3(TT / 64, HH, BB), dim3(128), 0, stream>>>(
      qkvb, 3 * CC, 0, qkvb, 3 * CC, CC, qkvb, 3 * CC, 2 * CC, yb, TT, TT, 1);
  k_gemm<<<dim3(CC / 64, BT / 128), dim3(256), 0, stream>>>(
      yb, wb_aproj, aproj_b, x, x1, BT, CC, CC, 2);

  // --- cross-attention to z ---
  k_layernorm<<<dim3(BT), dim3(256), 0, stream>>>(x1, lnc_w, lnc_b, hb);
  k_gemm<<<dim3(CC / 64, BT / 128), dim3(256), 0, stream>>>(
      hb, wb_cq, cq_b, nullptr, qb, BT, CC, CC, 0);
  k_gemm<<<dim3(2 * CC / 64, BSR / 128), dim3(256), 0, stream>>>(
      zb, wb_ckv, ckv_b, nullptr, kvb, BSR, 2 * CC, CC, 0);
  k_flash_attn<<<dim3(TT / 64, HH, BB), dim3(128), 0, stream>>>(
      qb, CC, 0, kvb, 2 * CC, 0, kvb, 2 * CC, CC, yb, TT, SS, 0);
  k_gemm<<<dim3(CC / 64, BT / 128), dim3(256), 0, stream>>>(
      yb, wb_cproj, cproj_b, x1, x2, BT, CC, CC, 2);

  // --- MLP ---
  k_layernorm<<<dim3(BT), dim3(256), 0, stream>>>(x2, ln2_w, ln2_b, hb);
  k_gemm<<<dim3(FC / 64, BT / 128), dim3(256), 0, stream>>>(
      hb, wb_fc, fc_b, nullptr, fcb, BT, FC, CC, 1);
  k_gemm<<<dim3(CC / 64, BT / 128), dim3(256), 0, stream>>>(
      fcb, wb_mproj, mproj_b, x2, out, BT, CC, FC, 2);
}
